// Edge_Conv_32890859553324
// MI455X (gfx1250) — compile-verified, matrix operand order checked by reading, not run on hardware
//
#include <hip/hip_runtime.h>

#define NB 16
#define PP 2048
#define CC 64
#define KN 16
#define C2 128
#define EPSBN 1e-5f

typedef __attribute__((ext_vector_type(16))) _Float16 v16h;
typedef __attribute__((ext_vector_type(8)))  _Float16 v8h;
typedef __attribute__((ext_vector_type(8)))  float    v8f;
typedef __attribute__((ext_vector_type(2)))  float    v2f;

union HV { v16h v; v8h h[2]; };

__device__ __forceinline__ v8f wmma_f16(v16h a, v16h b, v8f c) {
  return __builtin_amdgcn_wmma_f32_16x16x32_f16(false, a, false, b, (short)0, c, false, false);
}
__device__ __forceinline__ v8f wmma_f32k4(v2f a, v2f b, v8f c) {
  return __builtin_amdgcn_wmma_f32_16x16x4_f32(false, a, false, b, (short)0, c, false, false);
}

// ---------------------------------------------------------------------------
// Kernel 0: transpose features (N,C,P) -> X (N,P,C)
// ---------------------------------------------------------------------------
__global__ __launch_bounds__(256) void xpose_kernel(const float* __restrict__ feat,
                                                    float* __restrict__ X) {
  int i = blockIdx.x * 256 + threadIdx.x;          // over N*P*C
  int c = i & (CC - 1);
  int p = (i >> 6) & (PP - 1);
  int n = i >> 17;
  X[i] = feat[((size_t)n * CC + c) * PP + p];
}

// ---------------------------------------------------------------------------
// Kernel 1: kNN.  WG = one (n, 16-row p tile).  8 waves, each wave computes
// 16 q-tiles of 16x16 via v_wmma_f32_16x16x4_f32 (K loop = C/4 = 16 steps),
// writes s(q) = r(q) - 2*dot into a 16 x 2048 LDS slab, then per-row top-K
// selection by iterative wave-argmin extraction.
// ---------------------------------------------------------------------------
__global__ __launch_bounds__(256) void knn_kernel(const float* __restrict__ X,
                                                  int* __restrict__ knn_out) {
  __shared__ float r_lds[PP];
  __shared__ float dist[16 * PP];
  const int n    = blockIdx.x >> 7;            // P/16 = 128 tiles per batch
  const int p0   = (blockIdx.x & 127) << 4;
  const int tid  = threadIdx.x;
  const int lane = tid & 31, wave = tid >> 5;
  const int row  = lane & 15, half = lane >> 4;
  const float* Xn = X + (size_t)n * PP * CC;

  // r(q) = |x_q|^2 for all q of this batch
  for (int q = tid; q < PP; q += 256) {
    const float4* xr = (const float4*)(Xn + (size_t)q * CC);
    float s = 0.f;
    for (int j = 0; j < 16; ++j) {
      float4 v = xr[j];
      s += v.x * v.x + v.y * v.y + v.z * v.z + v.w * v.w;
    }
    r_lds[q] = s;
  }
  __syncthreads();

  // Hoist A fragments: rows p0..p0+15, f32 16x4 layout
  // (VGPR0: K=0 lanes0-15 / K=2 lanes16-31; VGPR1: K=1 / K=3)
  v2f afr[16];
  const float* xp = Xn + (size_t)(p0 + row) * CC;
  for (int kc = 0; kc < 16; ++kc) {
    v2f a;
    a.x = xp[kc * 4 + 2 * half];
    a.y = xp[kc * 4 + 2 * half + 1];
    afr[kc] = a;
  }

  for (int it = 0; it < 16; ++it) {
    const int qb = (wave * 16 + it) << 4;
    const float* xq = Xn + (size_t)(qb + row) * CC;   // B[k][col]: col = row
    v8f acc = {};
    for (int kc = 0; kc < 16; ++kc) {
      v2f b;
      b.x = xq[kc * 4 + 2 * half];
      b.y = xq[kc * 4 + 2 * half + 1];
      acc = wmma_f32k4(afr[kc], b, acc);
    }
    const int q  = qb + row;          // D col = lane&15
    const float rq = r_lds[q];
    for (int v = 0; v < 8; ++v) {     // D row = v + 8*half
      const int m = v + half * 8;
      float s = rq - 2.f * acc[v];
      if (q == p0 + m) s = 3.0e38f;   // exclude self
      dist[m * PP + q] = s;
    }
  }
  __syncthreads();

  // Selection: wave w owns rows 2w, 2w+1; 16 argmin extractions each.
  for (int rr = 0; rr < 2; ++rr) {
    const int m = wave * 2 + rr;
    volatile float* drow = dist + m * PP;
    int* orow = knn_out + ((size_t)n * PP + p0 + m) * KN;
    for (int k = 0; k < KN; ++k) {
      float best = 3.0e38f;
      int   bi   = 0x7fffffff;
      for (int j = 0; j < PP / 32; ++j) {
        const int c = lane + j * 32;
        const float dv = drow[c];
        if (dv < best || (dv == best && c < bi)) { best = dv; bi = c; }
      }
      for (int off = 16; off; off >>= 1) {
        const float ob = __shfl_xor(best, off, 32);
        const int   oi = __shfl_xor(bi, off, 32);
        if (ob < best || (ob == best && oi < bi)) { best = ob; bi = oi; }
      }
      if (lane == 0) { orow[k] = bi; drow[bi] = 3.0e38f; }
    }
  }
}

// ---------------------------------------------------------------------------
// Kernel 2: conv0.  Wave = one (n,p).  Build 16x128 edge tile [xc, xc-xj] in
// LDS (f16), B = w0 (64x128) in LDS, 4 k-chunks x 4 o-tiles of
// v_wmma_f32_16x16x32_f16.  Writes pre-BN y0 (f16) + channel sum/sumsq.
// ---------------------------------------------------------------------------
__global__ __launch_bounds__(256) void conv0_kernel(const float* __restrict__ X,
                                                    const int* __restrict__ knn_in,
                                                    const float* __restrict__ w0,
                                                    const float* __restrict__ b0,
                                                    _Float16* __restrict__ y0,
                                                    float* __restrict__ stats) {
  __shared__ _Float16 wB[CC * C2];            // [o][i], f16 copy of w0
  __shared__ float    sbias[CC];
  __shared__ float    s_sum[CC], s_sq[CC];
  __shared__ _Float16 Hs[8][KN * C2];         // per-wave edge tile
  const int tid = threadIdx.x, lane = tid & 31, wave = tid >> 5;

  for (int i = tid; i < CC * C2; i += 256) wB[i] = (_Float16)w0[i];
  for (int i = tid; i < CC; i += 256) { sbias[i] = b0[i]; s_sum[i] = 0.f; s_sq[i] = 0.f; }
  __syncthreads();

  const long gw = (long)blockIdx.x * 8 + wave;      // wave id == n*P + p
  const int  n  = (int)(gw >> 11);
  const int  p  = (int)(gw & (PP - 1));

  // Build H: lane (k = lane&15, hs = lane>>4) fills 32 channels of row k.
  {
    const int kk = lane & 15, hs = lane >> 4;
    const float* xc = X + ((size_t)n * PP + p) * CC + hs * 32;
    const int q = knn_in[((size_t)n * PP + p) * KN + kk];
    const float* xj = X + ((size_t)n * PP + q) * CC + hs * 32;
    _Float16* Hr = &Hs[wave][kk * C2];
    for (int c = 0; c < 32; c += 4) {
      float4 a = *(const float4*)(xc + c);
      float4 b = *(const float4*)(xj + c);
      Hr[hs * 32 + c + 0] = (_Float16)a.x;
      Hr[hs * 32 + c + 1] = (_Float16)a.y;
      Hr[hs * 32 + c + 2] = (_Float16)a.z;
      Hr[hs * 32 + c + 3] = (_Float16)a.w;
      Hr[64 + hs * 32 + c + 0] = (_Float16)(a.x - b.x);
      Hr[64 + hs * 32 + c + 1] = (_Float16)(a.y - b.y);
      Hr[64 + hs * 32 + c + 2] = (_Float16)(a.z - b.z);
      Hr[64 + hs * 32 + c + 3] = (_Float16)(a.w - b.w);
    }
  }
  __syncthreads();

  const int m = lane & 15, half = lane >> 4;
  // A-frags: 16-bit A layout: runs of 8 halves at K = kc*32+8*half and +16.
  v16h afr[4];
  for (int kc = 0; kc < 4; ++kc) {
    const _Float16* base = &Hs[wave][m * C2 + kc * 32];
    HV u;
    u.h[0] = *(const v8h*)(base + 8 * half);
    u.h[1] = *(const v8h*)(base + 16 + 8 * half);
    afr[kc] = u.v;
  }

  for (int ot = 0; ot < 4; ++ot) {
    const int o = ot * 16 + m;
    v8f acc;
    const float bv = sbias[o];
    for (int v = 0; v < 8; ++v) acc[v] = bv;
    for (int kc = 0; kc < 4; ++kc) {
      const _Float16* wr = &wB[o * C2 + kc * 32];
      HV u;
      u.h[0] = *(const v8h*)(wr + 8 * half);
      u.h[1] = *(const v8h*)(wr + 16 + 8 * half);
      acc = wmma_f16(afr[kc], u.v, acc);
    }
    _Float16* yrow = y0 + (((size_t)n * PP + p) * KN) * CC + o;
    float su = 0.f, sq = 0.f;
    for (int v = 0; v < 8; ++v) {
      const int mr = v + half * 8;
      const float val = acc[v];
      yrow[(size_t)mr * CC] = (_Float16)val;
      su += val; sq += val * val;
    }
    su += __shfl_xor(su, 16, 32);
    sq += __shfl_xor(sq, 16, 32);
    if (half == 0) { atomicAdd(&s_sum[o], su); atomicAdd(&s_sq[o], sq); }
  }
  __syncthreads();
  for (int i = tid; i < CC; i += 256) {
    atomicAdd(&stats[i], s_sum[i]);
    atomicAdd(&stats[CC + i], s_sq[i]);
  }
}

// ---------------------------------------------------------------------------
// Kernel 3/4: conv1 & conv2 (shared).  A from global f16 with per-channel
// scale/shift + ReLU applied on load.  2 k-chunks x 4 o-tiles of WMMA.
// ---------------------------------------------------------------------------
__global__ __launch_bounds__(256) void conv_mid_kernel(const _Float16* __restrict__ yin,
                                                       const float* __restrict__ w,
                                                       const float* __restrict__ b,
                                                       const float* __restrict__ bnscale,
                                                       const float* __restrict__ bnshift,
                                                       _Float16* __restrict__ yout,
                                                       float* __restrict__ stats) {
  __shared__ _Float16 wB[CC * CC];
  __shared__ float sbias[CC], s_scale[CC], s_shift[CC], s_sum[CC], s_sq[CC];
  const int tid = threadIdx.x, lane = tid & 31, wave = tid >> 5;
  for (int i = tid; i < CC * CC; i += 256) wB[i] = (_Float16)w[i];
  for (int i = tid; i < CC; i += 256) {
    sbias[i] = b[i]; s_scale[i] = bnscale[i]; s_shift[i] = bnshift[i];
    s_sum[i] = 0.f; s_sq[i] = 0.f;
  }
  __syncthreads();

  const long gw = (long)blockIdx.x * 8 + wave;   // == n*P + p
  const size_t np = (size_t)gw;
  const int m = lane & 15, half = lane >> 4;

  const _Float16* yrow = yin + (np * KN + m) * CC;
  v16h afr[2];
  for (int kc = 0; kc < 2; ++kc) {
    HV u;
    for (int r = 0; r < 2; ++r) {
      const int c0 = kc * 32 + r * 16 + 8 * half;
      v8h raw = *(const v8h*)(yrow + c0);
      v8h oh;
      for (int j = 0; j < 8; ++j) {
        float f = (float)raw[j] * s_scale[c0 + j] + s_shift[c0 + j];
        f = fmaxf(f, 0.f);
        oh[j] = (_Float16)f;
      }
      u.h[r] = oh;
    }
    afr[kc] = u.v;
  }

  for (int ot = 0; ot < 4; ++ot) {
    const int o = ot * 16 + m;
    v8f acc;
    const float bv = sbias[o];
    for (int v = 0; v < 8; ++v) acc[v] = bv;
    for (int kc = 0; kc < 2; ++kc) {
      const _Float16* wr = &wB[o * CC + kc * 32];
      HV u;
      u.h[0] = *(const v8h*)(wr + 8 * half);
      u.h[1] = *(const v8h*)(wr + 16 + 8 * half);
      acc = wmma_f16(afr[kc], u.v, acc);
    }
    _Float16* orow = yout + (np * KN) * CC + o;
    float su = 0.f, sq = 0.f;
    for (int v = 0; v < 8; ++v) {
      const int mr = v + half * 8;
      const float val = acc[v];
      orow[(size_t)mr * CC] = (_Float16)val;
      su += val; sq += val * val;
    }
    su += __shfl_xor(su, 16, 32);
    sq += __shfl_xor(sq, 16, 32);
    if (half == 0) { atomicAdd(&s_sum[o], su); atomicAdd(&s_sq[o], sq); }
  }
  __syncthreads();
  for (int i = tid; i < CC; i += 256) {
    atomicAdd(&stats[i], s_sum[i]);
    atomicAdd(&stats[CC + i], s_sq[i]);
  }
}

// ---------------------------------------------------------------------------
// Kernel 5: shortcut GEMM sc = X @ ws^T + bs (f16 WMMA), sc stored f32.
// Wave = 16 rows of flattened (n,p).
// ---------------------------------------------------------------------------
__global__ __launch_bounds__(256) void shortcut_kernel(const float* __restrict__ X,
                                                       const float* __restrict__ wsw,
                                                       const float* __restrict__ bs,
                                                       float* __restrict__ sc,
                                                       float* __restrict__ stats) {
  __shared__ _Float16 wB[CC * CC];
  __shared__ float sbias[CC], s_sum[CC], s_sq[CC];
  const int tid = threadIdx.x, lane = tid & 31, wave = tid >> 5;
  for (int i = tid; i < CC * CC; i += 256) wB[i] = (_Float16)wsw[i];
  for (int i = tid; i < CC; i += 256) { sbias[i] = bs[i]; s_sum[i] = 0.f; s_sq[i] = 0.f; }
  __syncthreads();

  const long gw = (long)blockIdx.x * 8 + wave;   // tile over N*P/16
  const size_t r0 = (size_t)gw * 16;
  const int m = lane & 15, half = lane >> 4;

  const float* xr = X + (r0 + m) * CC;
  v16h afr[2];
  for (int kc = 0; kc < 2; ++kc) {
    HV u;
    for (int r = 0; r < 2; ++r) {
      const int c0 = kc * 32 + r * 16 + 8 * half;
      float4 f0 = *(const float4*)(xr + c0);
      float4 f1 = *(const float4*)(xr + c0 + 4);
      v8h oh;
      oh[0] = (_Float16)f0.x; oh[1] = (_Float16)f0.y;
      oh[2] = (_Float16)f0.z; oh[3] = (_Float16)f0.w;
      oh[4] = (_Float16)f1.x; oh[5] = (_Float16)f1.y;
      oh[6] = (_Float16)f1.z; oh[7] = (_Float16)f1.w;
      u.h[r] = oh;
    }
    afr[kc] = u.v;
  }

  for (int ot = 0; ot < 4; ++ot) {
    const int o = ot * 16 + m;
    v8f acc;
    const float bv = sbias[o];
    for (int v = 0; v < 8; ++v) acc[v] = bv;
    for (int kc = 0; kc < 2; ++kc) {
      const _Float16* wr = &wB[o * CC + kc * 32];
      HV u;
      u.h[0] = *(const v8h*)(wr + 8 * half);
      u.h[1] = *(const v8h*)(wr + 16 + 8 * half);
      acc = wmma_f16(afr[kc], u.v, acc);
    }
    float su = 0.f, sq = 0.f;
    for (int v = 0; v < 8; ++v) {
      const size_t rrow = r0 + v + half * 8;
      const float val = acc[v];
      sc[rrow * CC + o] = val;
      su += val; sq += val * val;
    }
    su += __shfl_xor(su, 16, 32);
    sq += __shfl_xor(sq, 16, 32);
    if (half == 0) { atomicAdd(&s_sum[o], su); atomicAdd(&s_sq[o], sq); }
  }
  __syncthreads();
  for (int i = tid; i < CC; i += 256) {
    atomicAdd(&stats[i], s_sum[i]);
    atomicAdd(&stats[CC + i], s_sq[i]);
  }
}

// ---------------------------------------------------------------------------
// Kernel 6: fold BN stats -> per-channel scale/shift
// ---------------------------------------------------------------------------
__global__ void finalize_kernel(const float* __restrict__ stats,
                                const float* __restrict__ g,
                                const float* __restrict__ be,
                                float inv_cnt, float* __restrict__ ss) {
  const int i = threadIdx.x;
  if (i < CC) {
    const float mean = stats[i] * inv_cnt;
    const float var  = stats[CC + i] * inv_cnt - mean * mean;
    const float s    = g[i] * rsqrtf(var + EPSBN);
    ss[i]      = s;
    ss[CC + i] = be[i] - mean * s;
  }
}

// ---------------------------------------------------------------------------
// Kernel 7: out[n,o,p] = relu( mean_k relu(bn2(y2)) + bn_s(sc) )
// ---------------------------------------------------------------------------
__global__ __launch_bounds__(256) void final_kernel(const _Float16* __restrict__ y2,
                                                    const float* __restrict__ sc,
                                                    const float* __restrict__ ss2,
                                                    const float* __restrict__ sss,
                                                    float* __restrict__ out) {
  const int i = blockIdx.x * 256 + threadIdx.x;   // over N*C*P, p fastest
  const int p = i & (PP - 1);
  const int rest = i >> 11;
  const int o = rest & (CC - 1);
  const int n = rest >> 6;
  const float a = ss2[o], bsh = ss2[CC + o];
  const _Float16* yr = y2 + (((size_t)n * PP + p) * KN) * CC + o;
  float acc = 0.f;
  for (int k = 0; k < KN; ++k) {
    const float v = (float)yr[(size_t)k * CC] * a + bsh;
    acc += fmaxf(v, 0.f);
  }
  acc *= (1.f / KN);
  const float s = sc[((size_t)n * PP + p) * CC + o] * sss[o] + sss[CC + o];
  out[((size_t)n * CC + o) * PP + p] = fmaxf(acc + s, 0.f);
}

// ---------------------------------------------------------------------------
extern "C" void kernel_launch(void* const* d_in, const int* in_sizes, int n_in,
                              void* d_out, int out_size, void* d_ws, size_t ws_size,
                              hipStream_t stream) {
  const float* feat = (const float*)d_in[0];
  const float* w0  = (const float*)d_in[1];
  const float* b0  = (const float*)d_in[2];
  const float* g0  = (const float*)d_in[3];
  const float* be0 = (const float*)d_in[4];
  const float* w1  = (const float*)d_in[5];
  const float* b1  = (const float*)d_in[6];
  const float* g1  = (const float*)d_in[7];
  const float* be1 = (const float*)d_in[8];
  const float* w2  = (const float*)d_in[9];
  const float* b2  = (const float*)d_in[10];
  const float* g2  = (const float*)d_in[11];
  const float* be2 = (const float*)d_in[12];
  const float* wsw = (const float*)d_in[13];
  const float* bs  = (const float*)d_in[14];
  const float* gs  = (const float*)d_in[15];
  const float* bes = (const float*)d_in[16];

  char* base = (char*)d_ws;
  size_t off = 0;
  auto wsalloc = [&](size_t bytes) -> void* {
    void* p = base + off;
    off += (bytes + 255) & ~(size_t)255;
    return p;
  };
  float*     X     = (float*)wsalloc((size_t)NB * PP * CC * 4);          // 8 MB
  int*       knn   = (int*)wsalloc((size_t)NB * PP * KN * 4);            // 2 MB
  _Float16*  y0    = (_Float16*)wsalloc((size_t)NB * PP * KN * CC * 2);  // 64 MB
  _Float16*  y1    = (_Float16*)wsalloc((size_t)NB * PP * KN * CC * 2);  // 64 MB
  float*     scb   = (float*)wsalloc((size_t)NB * PP * CC * 4);          // 8 MB
  float*     stats = (float*)wsalloc(1024 * 4);                          // 4 stages sum/sq + scale/shift
  float* st0 = stats;        float* st1 = stats + 128;
  float* st2 = stats + 256;  float* stS = stats + 384;
  float* ss0 = stats + 512;  float* ss1 = stats + 640;
  float* ss2 = stats + 768;  float* ssS = stats + 896;
  _Float16* y2 = y0;  // reuse y0 buffer for conv2 output

  hipMemsetAsync(stats, 0, 512 * sizeof(float), stream);  // zero sum/sq stages

  xpose_kernel<<<(NB * PP * CC) / 256, 256, 0, stream>>>(feat, X);
  knn_kernel<<<NB * (PP / 16), 256, 0, stream>>>(X, knn);
  conv0_kernel<<<(NB * PP) / 8, 256, 0, stream>>>(X, knn, w0, b0, y0, st0);
  finalize_kernel<<<1, 64, 0, stream>>>(st0, g0, be0,
      1.f / ((float)NB * PP * KN), ss0);
  conv_mid_kernel<<<(NB * PP) / 8, 256, 0, stream>>>(y0, w1, b1, ss0, ss0 + CC, y1, st1);
  finalize_kernel<<<1, 64, 0, stream>>>(st1, g1, be1,
      1.f / ((float)NB * PP * KN), ss1);
  conv_mid_kernel<<<(NB * PP) / 8, 256, 0, stream>>>(y1, w2, b2, ss1, ss1 + CC, y2, st2);
  shortcut_kernel<<<(NB * PP / 16) / 8, 256, 0, stream>>>(X, wsw, bs, scb, stS);
  finalize_kernel<<<1, 64, 0, stream>>>(st2, g2, be2,
      1.f / ((float)NB * PP * KN), ss2);
  finalize_kernel<<<1, 64, 0, stream>>>(stS, gs, bes,
      1.f / ((float)NB * PP), ssS);
  final_kernel<<<(NB * CC * PP) / 256, 256, 0, stream>>>(y2, scb, ss2, ssS, (float*)d_out);
}
